// Attention_6966436954398
// MI455X (gfx1250) — matrix operand order repaired and checked by reading, not compiled, not run on hardware
//
#include <hip/hip_runtime.h>

// ---------------------------------------------------------------------------
// Flash-attention (causal) for B=2,H=16,S=2048,D=64 on gfx1250 (MI455X).
// f32 in/out, f16 WMMA (v_wmma_f32_16x16x32_f16) with f32 accumulation.
// Double-buffered async (global -> LDS) staging of K/V when available.
// ---------------------------------------------------------------------------

typedef __attribute__((ext_vector_type(16))) _Float16 v16h;
typedef __attribute__((ext_vector_type(8)))  float    v8f;
typedef __attribute__((ext_vector_type(4)))  float    f32x4;
typedef int v4i __attribute__((vector_size(4 * sizeof(int))));

#define WMMA_F16F32(a, b, c) \
    __builtin_amdgcn_wmma_f32_16x16x32_f16(false, (a), false, (b), (short)0, (c), false, false)

#if defined(__AMDGCN__) && __has_builtin(__builtin_amdgcn_global_load_async_to_lds_b128)
#define FA_ASYNC 1
#else
#define FA_ASYNC 0
#endif

#if FA_ASYNC
// Signature (from hipcc diagnostics): (global v4i*, lds v4i*, imm offset, imm cpol)
#define ASYNC_CPY_B128(gsrc, ldst)                                        \
    __builtin_amdgcn_global_load_async_to_lds_b128(                       \
        (__attribute__((address_space(1))) v4i*)(gsrc),                   \
        (__attribute__((address_space(3))) v4i*)(ldst), 0, 0)
#if __has_builtin(__builtin_amdgcn_s_wait_asynccnt)
#define WAIT_ASYNC0() __builtin_amdgcn_s_wait_asynccnt(0)
#else
#define WAIT_ASYNC0() asm volatile("s_wait_asynccnt 0" ::: "memory")
#endif
#endif

// ds_swizzle_b32 group-of-32 XOR patterns: offset = (xor<<10) | and(0x1f)
template <int PAT>
__device__ __forceinline__ float swz_f32(float x) {
    return __int_as_float(__builtin_amdgcn_ds_swizzle(__float_as_int(x), PAT));
}
__device__ __forceinline__ float redmax16(float v) {  // max across 16-lane half
    v = fmaxf(v, swz_f32<0x041f>(v));
    v = fmaxf(v, swz_f32<0x081f>(v));
    v = fmaxf(v, swz_f32<0x101f>(v));
    v = fmaxf(v, swz_f32<0x201f>(v));
    return v;
}
__device__ __forceinline__ float redsum16(float v) {  // sum across 16-lane half
    v += swz_f32<0x041f>(v);
    v += swz_f32<0x081f>(v);
    v += swz_f32<0x101f>(v);
    v += swz_f32<0x201f>(v);
    return v;
}

__device__ __forceinline__ v16h ld_v16h(const _Float16* p) {
    v16h r;
    __builtin_memcpy(&r, p, 32);   // 16B-aligned -> 2x ds_load_b128
    return r;
}
__device__ __forceinline__ f32x4 ld_f32x4(const float* p) {
    return *(const f32x4*)p;
}

__global__ __launch_bounds__(128)
void fa_causal_kernel(const float* __restrict__ Qg_,
                      const float* __restrict__ Kg_,
                      const float* __restrict__ Vg_,
                      float* __restrict__ Og_) {
    constexpr int S = 2048, D = 64;

    const int bh = blockIdx.y;                 // 0..31 (B*H)
    const int q0 = blockIdx.x * 64;            // block's first query row

    const float* Qg = Qg_ + (size_t)bh * S * D;
    const float* Kg = Kg_ + (size_t)bh * S * D;
    const float* Vg = Vg_ + (size_t)bh * S * D;
    float*       Og = Og_ + (size_t)bh * S * D;

    // f16 operand tiles (row strides are multiples of 8 halves -> 16B aligned)
    __shared__ __align__(16) _Float16 sK [32][72];    // K tile [key][dim]
    __shared__ __align__(16) _Float16 sVt[64][40];    // V tile transposed [dim][key]
    __shared__ __align__(16) _Float16 sP [4][16][40]; // per-wave P scratch [row][key]
#if FA_ASYNC
    // raw f32 double buffers fed by the async engine
    __shared__ __align__(16) float sRK[2][32][64];
    __shared__ __align__(16) float sRV[2][32][64];
#endif

    const int tid   = threadIdx.x;
    const int w     = tid >> 5;        // wave 0..3
    const int lane  = tid & 31;
    const int m16   = lane & 15;
    const int hi    = lane >> 4;
    const int base8 = hi * 8;

    const int qw0  = q0 + w * 16;      // wave's first query row
    const int qrow = qw0 + m16;        // this lane's A-matrix row

    // ---- Load Q tile into WMMA A-layout registers (scale folded in) -------
    const float qscale = 0.125f * 1.44269504088896340736f; // 1/sqrt(64)*log2(e)
    v16h qa[2];
#pragma unroll
    for (int c = 0; c < 2; ++c) {
        const float* qp = Qg + (size_t)qrow * D + c * 32 + hi * 16;
        f32x4 f0 = ld_f32x4(qp + 0);
        f32x4 f1 = ld_f32x4(qp + 4);
        f32x4 f2 = ld_f32x4(qp + 8);
        f32x4 f3 = ld_f32x4(qp + 12);
        v16h a;
#pragma unroll
        for (int j = 0; j < 4; ++j) {
            a[j]      = (_Float16)(f0[j] * qscale);
            a[4 + j]  = (_Float16)(f1[j] * qscale);
            a[8 + j]  = (_Float16)(f2[j] * qscale);
            a[12 + j] = (_Float16)(f3[j] * qscale);
        }
        qa[c] = a;
    }

    // ---- Accumulators and online-softmax row statistics -------------------
    v8f acc[4];
#pragma unroll
    for (int n = 0; n < 4; ++n) acc[n] = (v8f){};
    float mrow[8], lrow[8];
#pragma unroll
    for (int r = 0; r < 8; ++r) { mrow[r] = -__builtin_inff(); lrow[r] = 0.0f; }

    const int nkt = q0 / 32 + 2;       // tiles covering keys 0 .. q0+63
    const int kr  = tid >> 2;          // staging: key row 0..31
    const int db  = (tid & 3) * 16;    // staging: dim base 0/16/32/48

#if FA_ASYNC
    // Prologue: kick async copy of tile 0 into buffer 0 (4x b128 each for K,V).
    {
        const float* kp = Kg + (size_t)kr * D + db;
        const float* vp = Vg + (size_t)kr * D + db;
#pragma unroll
        for (int j = 0; j < 4; ++j) {
            ASYNC_CPY_B128(kp + 4 * j, &sRK[0][kr][db + 4 * j]);
            ASYNC_CPY_B128(vp + 4 * j, &sRV[0][kr][db + 4 * j]);
        }
    }
#endif

    for (int kt = 0; kt < nkt; ++kt) {
        const int kk0 = kt * 32;

#if FA_ASYNC
        WAIT_ASYNC0();        // this wave's copies for tile kt have landed
#endif
        __syncthreads();      // everyone's copies landed / prior reads done

#if FA_ASYNC
        // Issue async copy of tile kt+1 into the other buffer.
        if (kt + 1 < nkt) {
            const float* kp = Kg + (size_t)(kk0 + 32 + kr) * D + db;
            const float* vp = Vg + (size_t)(kk0 + 32 + kr) * D + db;
            float* dk = &sRK[(kt + 1) & 1][kr][db];
            float* dv = &sRV[(kt + 1) & 1][kr][db];
#pragma unroll
            for (int j = 0; j < 4; ++j) {
                ASYNC_CPY_B128(kp + 4 * j, dk + 4 * j);
                ASYNC_CPY_B128(vp + 4 * j, dv + 4 * j);
            }
        }
        const float* kp16 = &sRK[kt & 1][kr][db];
        const float* vp16 = &sRV[kt & 1][kr][db];
#else
        const float* kp16 = Kg + (size_t)(kk0 + kr) * D + db;
        const float* vp16 = Vg + (size_t)(kk0 + kr) * D + db;
        if (kt + 1 < nkt) {   // warm caches for next tile
            __builtin_prefetch(Kg + (size_t)(kk0 + 32 + kr) * D + db, 0, 0);
            __builtin_prefetch(Vg + (size_t)(kk0 + 32 + kr) * D + db, 0, 0);
        }
#endif
        // ---- Convert: f32 -> f16 operand tiles ----------------------------
        {
            f32x4 kf[4] = { ld_f32x4(kp16), ld_f32x4(kp16 + 4),
                            ld_f32x4(kp16 + 8), ld_f32x4(kp16 + 12) };
            _Float16 kh[16];
#pragma unroll
            for (int j = 0; j < 16; ++j) kh[j] = (_Float16)kf[j >> 2][j & 3];
            __builtin_memcpy(&sK[kr][db], kh, 32);

            f32x4 vf[4] = { ld_f32x4(vp16), ld_f32x4(vp16 + 4),
                            ld_f32x4(vp16 + 8), ld_f32x4(vp16 + 12) };
#pragma unroll
            for (int j = 0; j < 16; ++j)
                sVt[db + j][kr] = (_Float16)vf[j >> 2][j & 3];
        }

        __syncthreads();      // sK/sVt visible to all waves

        // ---- S = Q @ K^T for two 16-key subtiles --------------------------
        v8f sc[2];
#pragma unroll
        for (int t = 0; t < 2; ++t) {
            v16h b0 = ld_v16h(&sK[t * 16 + m16][hi * 16]);
            v16h b1 = ld_v16h(&sK[t * 16 + m16][32 + hi * 16]);
            v8f s = (v8f){};
            s = WMMA_F16F32(qa[0], b0, s);
            s = WMMA_F16F32(qa[1], b1, s);
            sc[t] = s;
        }

        // ---- Online softmax over the 16x32 score block --------------------
        const bool tile_masked = (kk0 + 31 > qw0);  // wave-uniform
#pragma unroll
        for (int r = 0; r < 8; ++r) {
            const int qr = qw0 + hi * 8 + r;
            float s0 = sc[0][r];
            float s1 = sc[1][r];
            if (tile_masked) {
                if (kk0 + m16 > qr)      s0 = -__builtin_inff();
                if (kk0 + 16 + m16 > qr) s1 = -__builtin_inff();
            }
            float vmax = redmax16(fmaxf(s0, s1));
            float mn   = fmaxf(mrow[r], vmax);
            float corr = __builtin_amdgcn_exp2f(mrow[r] - mn); // 0 on first tile
            float p0   = __builtin_amdgcn_exp2f(s0 - mn);      // 0 if masked
            float p1   = __builtin_amdgcn_exp2f(s1 - mn);
            float ps   = redsum16(p0 + p1);

            lrow[r] = lrow[r] * corr + ps;
            mrow[r] = mn;
#pragma unroll
            for (int n = 0; n < 4; ++n) acc[n][r] *= corr;

            sP[w][r + hi * 8][m16]      = (_Float16)p0;
            sP[w][r + hi * 8][16 + m16] = (_Float16)p1;
        }

        // sP is wave-private: per-wave DS ordering + compiler fence suffice.
        asm volatile("s_wait_dscnt 0" ::: "memory");

        // ---- O += P @ V ----------------------------------------------------
        v16h pa;
        {
            _Float16 ph[16];
            __builtin_memcpy(ph,     &sP[w][m16][hi * 16],      16);
            __builtin_memcpy(ph + 8, &sP[w][m16][hi * 16 + 8], 16);
            __builtin_memcpy(&pa, ph, 32);
        }
#pragma unroll
        for (int n = 0; n < 4; ++n) {
            v16h vb = ld_v16h(&sVt[n * 16 + m16][hi * 16]);
            acc[n] = WMMA_F16F32(pa, vb, acc[n]);
        }
    }

    // ---- Epilogue: normalize and store ------------------------------------
#pragma unroll
    for (int r = 0; r < 8; ++r) {
        const float inv = 1.0f / lrow[r];
        float* op = Og + (size_t)(qw0 + hi * 8 + r) * D;
#pragma unroll
        for (int n = 0; n < 4; ++n)
            op[n * 16 + m16] = acc[n][r] * inv;
    }
}

extern "C" void kernel_launch(void* const* d_in, const int* in_sizes, int n_in,
                              void* d_out, int out_size, void* d_ws, size_t ws_size,
                              hipStream_t stream) {
    (void)in_sizes; (void)n_in; (void)d_ws; (void)ws_size; (void)out_size;
    const float* Q = (const float*)d_in[0];
    const float* K = (const float*)d_in[1];
    const float* V = (const float*)d_in[2];
    // d_in[3] is the causal mask; the kernel applies causality analytically.
    float* O = (float*)d_out;

    constexpr int S = 2048, BH = 2 * 16;
    dim3 grid(S / 64, BH);
    fa_causal_kernel<<<grid, 128, 0, stream>>>(Q, K, V, O);
}
